// RWKV7CrossAttention_34815004902092
// MI455X (gfx1250) — matrix operand order repaired and checked by reading, not compiled, not run on hardware
//
#include <hip/hip_runtime.h>
#include <hip/hip_bf16.h>

// ---------------- problem constants (fixed by reference) ----------------
#define BB 2
#define TT 1024
#define CC 1024
#define HH 16
#define DD 64
#define MM (BB * TT)          // 2048 token rows
#define RW_ 64
#define RA_ 64
#define RV_ 16
#define RVP 32                // RV padded to WMMA K granularity
#define RG_ 128

static constexpr float W_DECAY  = -0.6065306597126334f;
static constexpr float NORM_EPS = 64.0f * 1e-5f;

typedef __attribute__((ext_vector_type(16))) _Float16 v16h;
typedef __attribute__((ext_vector_type(8)))  _Float16 v8h_t;
typedef __attribute__((ext_vector_type(8)))  float    v8f;

__device__ __forceinline__ float sigm(float x) { return 1.0f / (1.0f + expf(-x)); }

// ---------------------------------------------------------------------------
// 1) time-shift token mixing -> f16 activations (+ query convert)
// ---------------------------------------------------------------------------
__global__ void mix_kernel(const float* __restrict__ hs, const float* __restrict__ q,
                           const float* __restrict__ xw, const float* __restrict__ xk,
                           const float* __restrict__ xv, const float* __restrict__ xa,
                           const float* __restrict__ xg,
                           _Float16* __restrict__ xw16, _Float16* __restrict__ xk16,
                           _Float16* __restrict__ xv16, _Float16* __restrict__ xa16,
                           _Float16* __restrict__ xg16, _Float16* __restrict__ q16) {
  int idx = blockIdx.x * blockDim.x + threadIdx.x;
  if (idx >= MM * CC) return;
  int c = idx & (CC - 1);
  int t = (idx / CC) & (TT - 1);
  float h  = hs[idx];
  float sh = (t > 0) ? hs[idx - CC] : 0.0f;
  float d  = sh - h;
  xw16[idx] = (_Float16)(h + d * xw[c]);
  xk16[idx] = (_Float16)(h + d * xk[c]);
  xv16[idx] = (_Float16)(h + d * xv[c]);
  xa16[idx] = (_Float16)(h + d * xa[c]);
  xg16[idx] = (_Float16)(h + d * xg[c]);
  q16[idx]  = (_Float16)q[idx];
}

// ---------------------------------------------------------------------------
// 2) transpose + convert: src[K][N] f32 row-major -> dst[N][Kpad] f16 (zero pad)
// ---------------------------------------------------------------------------
__global__ void transpose_f32_f16(const float* __restrict__ src, _Float16* __restrict__ dst,
                                  int K, int N, int Kpad) {
  int k = blockIdx.x * blockDim.x + threadIdx.x;
  int n = blockIdx.y * blockDim.y + threadIdx.y;
  if (n >= N || k >= Kpad) return;
  dst[(size_t)n * Kpad + k] = (k < K) ? (_Float16)src[(size_t)k * N + n] : (_Float16)0.0f;
}

// ---------------------------------------------------------------------------
// 3) convert f32 -> f16 with optional activation + K padding (LoRA hidden)
//    act: 0=none, 1=tanh, 2=sigmoid
// ---------------------------------------------------------------------------
__global__ void convert_act_pad(const float* __restrict__ src, _Float16* __restrict__ dst,
                                int M, int R, int Rpad, int act) {
  int idx = blockIdx.x * blockDim.x + threadIdx.x;
  if (idx >= M * Rpad) return;
  int r = idx % Rpad;
  int m = idx / Rpad;
  float v = 0.0f;
  if (r < R) {
    v = src[(size_t)m * R + r];
    if (act == 1) v = tanhf(v);
    else if (act == 2) v = sigm(v);
  }
  dst[idx] = (_Float16)v;
}

// ---------------------------------------------------------------------------
// 4) WMMA GEMM: C[M,N] = A[M,K](f16, row-major, lda) * Bt[N,K](f16, ldb) ^T
//    One wave computes a 16 x (16*NT) strip, reusing its A fragment across NT
//    N-tiles (NT=4 -> 10 b128 loads per 4 WMMAs). 4 waves/block along M.
//    K % 32 == 0; N % (16*NT) == 0; M % 16 == 0.
// ---------------------------------------------------------------------------
template <int NT>
__global__ void gemm_wmma_f16(const _Float16* __restrict__ A, const _Float16* __restrict__ Bt,
                              float* __restrict__ C, int M, int N, int K,
                              int lda, int ldb, int ldc) {
  const int lane = threadIdx.x & 31;
  const int wave = threadIdx.x >> 5;
  const int m0   = (blockIdx.x * 4 + wave) * 16;
  const int n0   = blockIdx.y * 16 * NT;
  if (m0 >= M) return;

  const int hi  = (lane >= 16) ? 1 : 0;
  const int idx = lane & 15;
  // A layout (16-bit, 16x32): lane<16 -> K {0..7,16..23}; lane>=16 -> +8
  const _Float16* ap = A + (size_t)(m0 + idx) * lda + (hi ? 8 : 0);
  // B layout (16-bit, 32x16): lane = N col; lane<16 -> K {0..7,16..23}; lane>=16 -> +8
  const _Float16* bp[NT];
#pragma unroll
  for (int nt = 0; nt < NT; ++nt)
    bp[nt] = Bt + (size_t)(n0 + nt * 16 + idx) * ldb + (hi ? 8 : 0);

  v8f acc[NT];
#pragma unroll
  for (int nt = 0; nt < NT; ++nt) acc[nt] = (v8f){};

  for (int kk = 0; kk < K; kk += 32) {
    v8h_t alo = *(const v8h_t*)(ap + kk);
    v8h_t ahi = *(const v8h_t*)(ap + kk + 16);
    v16h a;
#pragma unroll
    for (int i = 0; i < 8; ++i) { a[i] = alo[i]; a[i + 8] = ahi[i]; }
#pragma unroll
    for (int nt = 0; nt < NT; ++nt) {
      v8h_t blo = *(const v8h_t*)(bp[nt] + kk);
      v8h_t bhi = *(const v8h_t*)(bp[nt] + kk + 16);
      v16h b;
#pragma unroll
      for (int i = 0; i < 8; ++i) { b[i] = blo[i]; b[i + 8] = bhi[i]; }
      acc[nt] = __builtin_amdgcn_wmma_f32_16x16x32_f16(false, a, false, b, (short)0,
                                                       acc[nt], false, false);
    }
  }
  // C/D layout: VGPR g -> row m0 + g + 8*hi, col n0 + idx
#pragma unroll
  for (int nt = 0; nt < NT; ++nt) {
    float* cp = C + (size_t)(m0 + 8 * hi) * ldc + n0 + nt * 16 + idx;
#pragma unroll
    for (int g = 0; g < 8; ++g) cp[(size_t)g * ldc] = acc[nt][g];
  }
}

// ---------------------------------------------------------------------------
// 5) per-(b,t,h) fusion: decay exp, a-gate, v lerp, k addcmul, kk normalize
//    32 threads per block, 2 channels per thread. In-place aliasing safe:
//    each thread reads then writes only its own elements.
// ---------------------------------------------------------------------------
__global__ void headfuse_kernel(const float* __restrict__ ktmp,   // k GEMM out (overwritten -> k_fin)
                                const float* __restrict__ a2,     // a lora out (overwritten -> bb)
                                const float* __restrict__ w2,     // w lora out (overwritten -> aa)
                                const float* __restrict__ v2,     // v lora out (overwritten -> ew)
                                const float* __restrict__ vtmp,   // v GEMM out (overwritten -> v_fin)
                                const float* __restrict__ v_first,
                                const float* __restrict__ a_lb, const float* __restrict__ w_lb,
                                const float* __restrict__ v_lb,
                                const float* __restrict__ k_k, const float* __restrict__ k_a,
                                float* __restrict__ kfin, float* __restrict__ bb,
                                float* __restrict__ aa, float* __restrict__ ew,
                                float* __restrict__ vfin) {
  const int bth  = blockIdx.x;          // (b*T + t)*H + h
  const int h    = bth & (HH - 1);
  const int lane = threadIdx.x;
  const size_t base = (size_t)bth * DD;

  float kt[2], kkv[2];
#pragma unroll
  for (int e = 0; e < 2; ++e) {
    int d = lane + 32 * e;
    kt[e]  = ktmp[base + d];
    kkv[e] = kt[e] * k_k[h * DD + d];
  }
  float ss = kkv[0] * kkv[0] + kkv[1] * kkv[1];
#pragma unroll
  for (int off = 16; off > 0; off >>= 1) ss += __shfl_xor(ss, off, 32);
  const float inv = 1.0f / fmaxf(sqrtf(ss), 1e-12f);

#pragma unroll
  for (int e = 0; e < 2; ++e) {
    int d = lane + 32 * e;
    int c = h * DD + d;
    size_t idx = base + d;
    float av  = sigm(a2[idx] + a_lb[c]);
    float kkn = kkv[e] * inv;
    float wl  = W_DECAY * sigm(w2[idx] + w_lb[c]);
    float sv  = sigm(v2[idx] + v_lb[c]);
    float vt  = vtmp[idx];
    aa[idx]   = -kkn;
    bb[idx]   = kkn * av;
    kfin[idx] = kt[e] * (1.0f + (av - 1.0f) * k_a[c]);
    ew[idx]   = expf(wl);
    vfin[idx] = vt + sv * (v_first[idx] - vt);
  }
}

// ---------------------------------------------------------------------------
// 6) RWKV7 sequential scan. One block per (b,h); 256 threads = 8 waves.
//    Thread (vrow, kq) owns S[vrow][kq*16 .. kq*16+15] in registers.
// ---------------------------------------------------------------------------
__global__ void __launch_bounds__(256) scan_kernel(
    const float* __restrict__ r, const float* __restrict__ ew,
    const float* __restrict__ k, const float* __restrict__ v,
    const float* __restrict__ aa, const float* __restrict__ bb,
    float* __restrict__ o) {
  const int bh = blockIdx.x;            // b*H + h
  const int b  = bh / HH;
  const int h  = bh & (HH - 1);
  const int tid  = threadIdx.x;
  const int vrow = tid & 63;
  const int kq   = tid >> 6;            // 0..3

  __shared__ float lr[DD], lw[DD], lk[DD], la[DD], lb[DD], lv[DD];
  __shared__ float red[4][DD];

  float s[16];
#pragma unroll
  for (int j = 0; j < 16; ++j) s[j] = 0.0f;

  const size_t stride = (size_t)HH * DD;                 // per time step
  size_t off = ((size_t)b * TT * HH + h) * DD;           // t = 0

  for (int t = 0; t < TT; ++t, off += stride) {
    if (kq == 0)      { lr[vrow] = r[off + vrow];  lw[vrow] = ew[off + vrow]; }
    else if (kq == 1) { lk[vrow] = k[off + vrow];  la[vrow] = aa[off + vrow]; }
    else if (kq == 2) { lb[vrow] = bb[off + vrow]; lv[vrow] = v[off + vrow]; }
    __syncthreads();

    // sa = S @ a  (reduce over k)
    float p = 0.0f;
#pragma unroll
    for (int j = 0; j < 16; ++j) p += s[j] * la[kq * 16 + j];
    red[kq][vrow] = p;
    __syncthreads();
    const float sa = red[0][vrow] + red[1][vrow] + red[2][vrow] + red[3][vrow];
    const float vt = lv[vrow];
    __syncthreads();

    // S = S*diag(ew) + sa b^T + v k^T ;  p2 = partial of S @ r
    float p2 = 0.0f;
#pragma unroll
    for (int j = 0; j < 16; ++j) {
      const int kc = kq * 16 + j;
      float sv = fmaf(s[j], lw[kc], fmaf(sa, lb[kc], vt * lk[kc]));
      s[j] = sv;
      p2 += sv * lr[kc];
    }
    red[kq][vrow] = p2;
    __syncthreads();
    if (kq == 0)
      o[off + vrow] = red[0][vrow] + red[1][vrow] + red[2][vrow] + red[3][vrow];
  }
}

// ---------------------------------------------------------------------------
// 7) GroupNorm + bonus term + gate -> f16 rows for final GEMM
// ---------------------------------------------------------------------------
__global__ void post_kernel(const float* __restrict__ o, const float* __restrict__ rbuf,
                            const float* __restrict__ kfin, const float* __restrict__ vfin,
                            const float* __restrict__ r_k, const float* __restrict__ gn_w,
                            const float* __restrict__ gn_b, const float* __restrict__ g2,
                            _Float16* __restrict__ og16) {
  const int bth  = blockIdx.x;
  const int h    = bth & (HH - 1);
  const int lane = threadIdx.x;
  const size_t base = (size_t)bth * DD;

  float ov[2], dotp = 0.0f, sum = 0.0f, sq = 0.0f;
#pragma unroll
  for (int e = 0; e < 2; ++e) {
    int d = lane + 32 * e;
    ov[e] = o[base + d];
    sum += ov[e];
    sq  += ov[e] * ov[e];
    dotp += rbuf[base + d] * kfin[base + d] * r_k[h * DD + d];
  }
#pragma unroll
  for (int off = 16; off > 0; off >>= 1) {
    sum  += __shfl_xor(sum, off, 32);
    sq   += __shfl_xor(sq, off, 32);
    dotp += __shfl_xor(dotp, off, 32);
  }
  const float mu   = sum * (1.0f / DD);
  const float var  = sq * (1.0f / DD) - mu * mu;
  const float rstd = rsqrtf(var + NORM_EPS);

#pragma unroll
  for (int e = 0; e < 2; ++e) {
    int d = lane + 32 * e;
    int c = h * DD + d;
    size_t idx = base + d;
    float y = (ov[e] - mu) * rstd * gn_w[c] + gn_b[c] + dotp * vfin[idx];
    og16[idx] = (_Float16)(y * g2[idx]);
  }
}

// ---------------------------------------------------------------------------
// host-side launcher
// ---------------------------------------------------------------------------
static inline size_t al256(size_t x) { return (x + 255) & ~(size_t)255; }

extern "C" void kernel_launch(void* const* d_in, const int* in_sizes, int n_in,
                              void* d_out, int out_size, void* d_ws, size_t ws_size,
                              hipStream_t stream) {
  (void)in_sizes; (void)n_in; (void)out_size; (void)ws_size;
  const float* query   = (const float*)d_in[0];
  const float* hidden  = (const float*)d_in[1];
  const float* v_first = (const float*)d_in[2];
  const float* x_w = (const float*)d_in[4];
  const float* x_k = (const float*)d_in[5];
  const float* x_v = (const float*)d_in[6];
  const float* x_a = (const float*)d_in[7];
  const float* x_g = (const float*)d_in[8];
  const float* k_k = (const float*)d_in[9];
  const float* k_a = (const float*)d_in[10];
  const float* r_k = (const float*)d_in[11];
  const float* W_r = (const float*)d_in[12];
  const float* W_k = (const float*)d_in[13];
  const float* W_v = (const float*)d_in[14];
  const float* W_o = (const float*)d_in[15];
  const float* w_lA = (const float*)d_in[16];
  const float* w_lB = (const float*)d_in[17];
  const float* w_lb = (const float*)d_in[18];
  const float* v_lA = (const float*)d_in[19];
  const float* v_lB = (const float*)d_in[20];
  const float* v_lb = (const float*)d_in[21];
  const float* a_lA = (const float*)d_in[22];
  const float* a_lB = (const float*)d_in[23];
  const float* a_lb = (const float*)d_in[24];
  const float* g_lA = (const float*)d_in[25];
  const float* g_lB = (const float*)d_in[26];
  const float* gn_w = (const float*)d_in[27];
  const float* gn_b = (const float*)d_in[28];
  float* out = (float*)d_out;

  // ---- workspace carve-up -------------------------------------------------
  char* ws = (char*)d_ws;
  size_t off = 0;
  auto grabH = [&](size_t elems) { _Float16* p = (_Float16*)(ws + off); off = al256(off + elems * sizeof(_Float16)); return p; };
  auto grabF = [&](size_t elems) { float* p = (float*)(ws + off); off = al256(off + elems * sizeof(float)); return p; };

  const size_t MC = (size_t)MM * CC;
  _Float16* q16  = grabH(MC);
  _Float16* xw16 = grabH(MC);
  _Float16* xk16 = grabH(MC);
  _Float16* xv16 = grabH(MC);
  _Float16* xa16 = grabH(MC);
  _Float16* xg16 = grabH(MC);
  _Float16* og16 = grabH(MC);

  _Float16* Wrt = grabH((size_t)CC * CC);
  _Float16* Wkt = grabH((size_t)CC * CC);
  _Float16* Wvt = grabH((size_t)CC * CC);
  _Float16* Wot = grabH((size_t)CC * CC);
  _Float16* wlAt = grabH((size_t)RW_ * CC);
  _Float16* alAt = grabH((size_t)RA_ * CC);
  _Float16* vlAt = grabH((size_t)RV_ * CC);
  _Float16* glAt = grabH((size_t)RG_ * CC);
  _Float16* wlBt = grabH((size_t)CC * RW_);
  _Float16* alBt = grabH((size_t)CC * RA_);
  _Float16* vlBt = grabH((size_t)CC * RVP);
  _Float16* glBt = grabH((size_t)CC * RG_);

  _Float16* wh16 = grabH((size_t)MM * RW_);
  _Float16* ah16 = grabH((size_t)MM * RA_);
  _Float16* vh16 = grabH((size_t)MM * RVP);
  _Float16* gh16 = grabH((size_t)MM * RG_);

  float* rbuf = grabF(MC);   // r = q @ W_r
  float* kbuf = grabF(MC);   // k GEMM out, then k_fin (in-place)
  float* vbuf = grabF(MC);   // v GEMM out, then v_fin (in-place)
  float* wbuf = grabF(MC);   // w lora out, then aa (in-place)
  float* abuf = grabF(MC);   // a lora out, then bb (in-place)
  float* ebuf = grabF(MC);   // v lora out, then exp(w) (in-place)
  float* gbuf = grabF(MC);   // g lora out (final gate)
  float* obuf = grabF(MC);   // scan output
  float* wh1  = grabF((size_t)MM * RW_);
  float* ah1  = grabF((size_t)MM * RA_);
  float* vh1  = grabF((size_t)MM * RV_);
  float* gh1  = grabF((size_t)MM * RG_);

  // ---- 1) token mixing + f16 conversion -----------------------------------
  mix_kernel<<<dim3((MM * CC + 255) / 256), dim3(256), 0, stream>>>(
      hidden, query, x_w, x_k, x_v, x_a, x_g, xw16, xk16, xv16, xa16, xg16, q16);

  // ---- 2) weight transposes (f32 [K,N] -> f16 [N,Kpad]) -------------------
  dim3 tb(32, 8);
  auto tgrid = [](int Kpad, int N) { return dim3((Kpad + 31) / 32, (N + 7) / 8); };
  transpose_f32_f16<<<tgrid(CC, CC), tb, 0, stream>>>(W_r, Wrt, CC, CC, CC);
  transpose_f32_f16<<<tgrid(CC, CC), tb, 0, stream>>>(W_k, Wkt, CC, CC, CC);
  transpose_f32_f16<<<tgrid(CC, CC), tb, 0, stream>>>(W_v, Wvt, CC, CC, CC);
  transpose_f32_f16<<<tgrid(CC, CC), tb, 0, stream>>>(W_o, Wot, CC, CC, CC);
  transpose_f32_f16<<<tgrid(CC, RW_), tb, 0, stream>>>(w_lA, wlAt, CC, RW_, CC);
  transpose_f32_f16<<<tgrid(CC, RA_), tb, 0, stream>>>(a_lA, alAt, CC, RA_, CC);
  transpose_f32_f16<<<tgrid(CC, RV_), tb, 0, stream>>>(v_lA, vlAt, CC, RV_, CC);
  transpose_f32_f16<<<tgrid(CC, RG_), tb, 0, stream>>>(g_lA, glAt, CC, RG_, CC);
  transpose_f32_f16<<<tgrid(RW_, CC), tb, 0, stream>>>(w_lB, wlBt, RW_, CC, RW_);
  transpose_f32_f16<<<tgrid(RA_, CC), tb, 0, stream>>>(a_lB, alBt, RA_, CC, RA_);
  transpose_f32_f16<<<tgrid(RVP, CC), tb, 0, stream>>>(v_lB, vlBt, RV_, CC, RVP);
  transpose_f32_f16<<<tgrid(RG_, CC), tb, 0, stream>>>(g_lB, glBt, RG_, CC, RG_);

  // ---- 3) big GEMMs + LoRA stage 1 (all WMMA) -----------------------------
  dim3 gblk(128);
  auto ggrid4 = [](int M, int N) { return dim3((M + 63) / 64, N / 64); };
  auto ggrid1 = [](int M, int N) { return dim3((M + 63) / 64, N / 16); };
  gemm_wmma_f16<4><<<ggrid4(MM, CC), gblk, 0, stream>>>(q16,  Wrt, rbuf, MM, CC, CC, CC, CC, CC);
  gemm_wmma_f16<4><<<ggrid4(MM, CC), gblk, 0, stream>>>(xk16, Wkt, kbuf, MM, CC, CC, CC, CC, CC);
  gemm_wmma_f16<4><<<ggrid4(MM, CC), gblk, 0, stream>>>(xv16, Wvt, vbuf, MM, CC, CC, CC, CC, CC);
  gemm_wmma_f16<4><<<ggrid4(MM, RW_), gblk, 0, stream>>>(xw16, wlAt, wh1, MM, RW_, CC, CC, CC, RW_);
  gemm_wmma_f16<4><<<ggrid4(MM, RA_), gblk, 0, stream>>>(xa16, alAt, ah1, MM, RA_, CC, CC, CC, RA_);
  gemm_wmma_f16<1><<<ggrid1(MM, RV_), gblk, 0, stream>>>(xv16, vlAt, vh1, MM, RV_, CC, CC, CC, RV_);
  gemm_wmma_f16<4><<<ggrid4(MM, RG_), gblk, 0, stream>>>(xg16, glAt, gh1, MM, RG_, CC, CC, CC, RG_);

  // ---- 4) LoRA hidden activation + f16 (padded) ---------------------------
  convert_act_pad<<<dim3((MM * RW_ + 255) / 256), dim3(256), 0, stream>>>(wh1, wh16, MM, RW_, RW_, 1);
  convert_act_pad<<<dim3((MM * RA_ + 255) / 256), dim3(256), 0, stream>>>(ah1, ah16, MM, RA_, RA_, 0);
  convert_act_pad<<<dim3((MM * RVP + 255) / 256), dim3(256), 0, stream>>>(vh1, vh16, MM, RV_, RVP, 0);
  convert_act_pad<<<dim3((MM * RG_ + 255) / 256), dim3(256), 0, stream>>>(gh1, gh16, MM, RG_, RG_, 2);

  // ---- 5) LoRA stage 2 (WMMA) ---------------------------------------------
  gemm_wmma_f16<4><<<ggrid4(MM, CC), gblk, 0, stream>>>(wh16, wlBt, wbuf, MM, CC, RW_, RW_, RW_, CC);
  gemm_wmma_f16<4><<<ggrid4(MM, CC), gblk, 0, stream>>>(ah16, alBt, abuf, MM, CC, RA_, RA_, RA_, CC);
  gemm_wmma_f16<4><<<ggrid4(MM, CC), gblk, 0, stream>>>(vh16, vlBt, ebuf, MM, CC, RVP, RVP, RVP, CC);
  gemm_wmma_f16<4><<<ggrid4(MM, CC), gblk, 0, stream>>>(gh16, glBt, gbuf, MM, CC, RG_, RG_, RG_, CC);

  // ---- 6) per-head fusion (in-place aliasing) -----------------------------
  headfuse_kernel<<<dim3(MM * HH), dim3(32), 0, stream>>>(
      kbuf, abuf, wbuf, ebuf, vbuf, v_first, a_lb, w_lb, v_lb, k_k, k_a,
      /*kfin=*/kbuf, /*bb=*/abuf, /*aa=*/wbuf, /*ew=*/ebuf, /*vfin=*/vbuf);

  // ---- 7) sequential RWKV7 scan -------------------------------------------
  scan_kernel<<<dim3(BB * HH), dim3(256), 0, stream>>>(rbuf, ebuf, kbuf, vbuf, wbuf, abuf, obuf);

  // ---- 8) GroupNorm + bonus + gate ----------------------------------------
  post_kernel<<<dim3(MM * HH), dim3(32), 0, stream>>>(
      obuf, rbuf, kbuf, vbuf, r_k, gn_w, gn_b, gbuf, og16);

  // ---- 9) final projection (WMMA) -----------------------------------------
  gemm_wmma_f16<4><<<ggrid4(MM, CC), gblk, 0, stream>>>(og16, Wot, out, MM, CC, CC, CC, CC, CC);
}
